// SudokuRRN_64330020159588
// MI455X (gfx1250) — compile-verified, matrix operand order checked by reading, not run on hardware
//
#include <hip/hip_runtime.h>
#include <hip/hip_bf16.h>

typedef __bf16 bf16_t;
typedef bf16_t v16bf __attribute__((ext_vector_type(16)));
typedef bf16_t v8bf  __attribute__((ext_vector_type(8)));
typedef float  v8f   __attribute__((ext_vector_type(8)));

#define HID 192
#define NSTEPS 16
#define GCOLS 96          // columns of B staged per block (6 WMMA tiles)
#define NT (GCOLS / 16)   // 6 column tiles per block
#define RS 2              // row strips (of 16) per wave -> 32 rows/wave
#define LDS_LD 200        // padded LDS row stride (bf16 elems) to spread banks

// ---------------------------------------------------------------------------
// WMMA fragment load: 16x32 bf16 tile from a row-major (rows x ld) matrix.
// CDNA5 layout (ISA 7.12.2): lanes 0-15 hold row M=lane with K in
// {0..7} U {16..23}; lanes 16-31 hold row M=lane-16 with K in {8..15} U {24..31}.
// Two 16-byte loads per lane (global_load_b128 or ds_load_b128).
// ---------------------------------------------------------------------------
__device__ inline v16bf load_frag16x32(const bf16_t* __restrict__ base, int ld,
                                       int rowBase, int kBase, int lane) {
  int r  = rowBase + (lane & 15);
  int k0 = kBase + ((lane & 16) ? 8 : 0);
  const bf16_t* p = base + (size_t)r * ld + k0;
  v8bf lo = *reinterpret_cast<const v8bf*>(p);
  v8bf hi = *reinterpret_cast<const v8bf*>(p + 16);
  v16bf out;
#pragma unroll
  for (int i = 0; i < 8; ++i) { out[i] = lo[i]; out[i + 8] = hi[i]; }
  return out;
}

// ---------------------------------------------------------------------------
// GEMM: out(M x Ncols) = A(M x 192, bf16) @ BT(Ncols x 192, bf16)^T [+ bias]
// K fixed at 192 (6 chunks of 32). Grid: (M/(8*32), Ncols/GCOLS).
// Block = 8 waves; B slice (GCOLS x 192) cooperatively staged in LDS, shared
// by all waves. Each wave owns TWO 16-row strips: every B fragment fetched
// from LDS feeds two back-to-back WMMAs (one per strip), so the second WMMA
// issues with no wait and LDS traffic per FLOP is halved.
// Epilogue is branch-free (compile-time flags).
// ---------------------------------------------------------------------------
template <bool HASBIAS, bool OUTF, bool OUTB>
__global__ __launch_bounds__(256) void gemm_k192_wmma(
    const bf16_t* __restrict__ A, const bf16_t* __restrict__ BT,
    const float* __restrict__ bias, float* __restrict__ outF,
    bf16_t* __restrict__ outB, int M, int Ncols) {
  __shared__ bf16_t sB[GCOLS * LDS_LD];
  const int tid = threadIdx.x;
  const int lane = tid & 31;
  const int wave = tid >> 5;
  const int colBase = blockIdx.y * GCOLS;

  // ---- cooperative stage of B slice (GCOLS x 192 bf16) into LDS ----
  {
    const uint4* src = reinterpret_cast<const uint4*>(BT + (size_t)colBase * HID);
    // 24 x uint4 (384B) per row; GCOLS rows
#pragma unroll 3
    for (int i = tid; i < GCOLS * 24; i += 256) {
      int r = i / 24, s = i % 24;
      uint4 v = src[r * 24 + s];
      *reinterpret_cast<uint4*>(&sB[r * LDS_LD + s * 8]) = v;
    }
  }
  __syncthreads();

  const int mb = (blockIdx.x * 8 + wave) * (16 * RS);
  if (mb >= M) return;

  // A fragments for both strips: global load clauses, reused across 6 tiles
  v16bf afrag[RS][6];
#pragma unroll
  for (int s = 0; s < RS; ++s)
#pragma unroll
    for (int kb = 0; kb < 6; ++kb)
      afrag[s][kb] = load_frag16x32(A, HID, mb + s * 16, kb * 32, lane);

  v8f acc[RS][NT] = {};
#pragma unroll
  for (int kb = 0; kb < 6; ++kb) {
#pragma unroll
    for (int tt = 0; tt < NT; ++tt) {
      v16bf bfrag = load_frag16x32(sB, LDS_LD, tt * 16, kb * 32, lane);
      // one LDS fetch -> two WMMAs (second needs no wait)
#pragma unroll
      for (int s = 0; s < RS; ++s)
        acc[s][tt] = __builtin_amdgcn_wmma_f32_16x16x32_bf16(
            false, afrag[s][kb], false, bfrag, (short)0, acc[s][tt], false, false);
    }
  }

  // D layout: lane -> column n; VGPR v -> row v (lanes 0-15) / v+8 (16-31)
#pragma unroll
  for (int tt = 0; tt < NT; ++tt) {
    const int n = colBase + tt * 16 + (lane & 15);
    const float bv = HASBIAS ? bias[n] : 0.0f;
#pragma unroll
    for (int s = 0; s < RS; ++s) {
      const int mRow = mb + s * 16 + ((lane & 16) ? 8 : 0);
#pragma unroll
      for (int v = 0; v < 8; ++v) {
        float val = acc[s][tt][v] + bv;
        size_t idx = (size_t)(mRow + v) * Ncols + n;
        if (OUTF) outF[idx] = val;
        if (OUTB) outB[idx] = (bf16_t)val;
      }
    }
  }
}

// ---------------------------------------------------------------------------
// Weight prep: bf16 conversion + transposes so all GEMM B operands are
// (Nout x K) row-major.  WT_m1 packs [W_m1_top^T | W_m1_bot^T] (384 x 192).
// W_ih / W_hh are used as agg @ W^T, so their native (576 x 192) layout is
// already the transposed form -> straight convert.  b_m2s = 20 * b_m2.
// ---------------------------------------------------------------------------
__global__ void prep_weights(const float* __restrict__ W_m1, const float* __restrict__ W_m2,
                             const float* __restrict__ W_ih, const float* __restrict__ W_hh,
                             const float* __restrict__ b_m2,
                             bf16_t* __restrict__ WT_m1, bf16_t* __restrict__ WT_m2,
                             bf16_t* __restrict__ Wih_bf, bf16_t* __restrict__ Whh_bf,
                             float* __restrict__ b_m2s) {
  int i = blockIdx.x * blockDim.x + threadIdx.x;
  if (i < 384 * HID) {
    int j = i / HID, k = i % HID;
    float v = (j < HID) ? W_m1[(size_t)k * HID + j]
                        : W_m1[(size_t)(HID + k) * HID + (j - HID)];
    WT_m1[i] = (bf16_t)v;
  }
  if (i < HID * HID) {
    int j = i / HID, k = i % HID;
    WT_m2[i] = (bf16_t)W_m2[(size_t)k * HID + j];
  }
  if (i < 576 * HID) {
    Wih_bf[i] = (bf16_t)W_ih[i];
    Whh_bf[i] = (bf16_t)W_hh[i];
  }
  if (i < HID) b_m2s[i] = 20.0f * b_m2[i];
}

// --------------------------- block-wide reduction --------------------------
__device__ inline float block_sum192(float* sh, float v, int c) {
  sh[c] = v; __syncthreads();
  if (c < 96) sh[c] += sh[c + 96]; __syncthreads();
  if (c < 48) sh[c] += sh[c + 48]; __syncthreads();
  if (c < 24) sh[c] += sh[c + 24]; __syncthreads();
  if (c < 12) sh[c] += sh[c + 12]; __syncthreads();
  if (c <  6) sh[c] += sh[c +  6]; __syncthreads();
  if (c <  3) sh[c] += sh[c +  3]; __syncthreads();
  float s = sh[0] + sh[1] + sh[2];
  __syncthreads();
  return s;
}

// ---------------------------------------------------------------------------
// h0 = LN(x @ W_in + b_in) + pos[n % 81]   (x is one-hot over 10)
// ---------------------------------------------------------------------------
__global__ __launch_bounds__(192) void init_h(
    const float* __restrict__ x, const float* __restrict__ W_in,
    const float* __restrict__ b_in, const float* __restrict__ g_in,
    const float* __restrict__ be_in, const float* __restrict__ pos,
    float* __restrict__ h, bf16_t* __restrict__ h_bf) {
  __shared__ float sh[HID];
  int n = blockIdx.x, c = threadIdx.x;
  const float* xr = x + (size_t)n * 10;
  float acc = b_in[c];
#pragma unroll
  for (int d = 0; d < 10; ++d) acc += xr[d] * W_in[d * HID + c];
  float mean = block_sum192(sh, acc, c) * (1.0f / HID);
  float dv = acc - mean;
  float var = block_sum192(sh, dv * dv, c) * (1.0f / HID);
  float v = dv * rsqrtf(var + 1e-5f) * g_in[c] + be_in[c] + pos[(size_t)(n % 81) * HID + c];
  h[(size_t)n * HID + c] = v;
  h_bf[(size_t)n * HID + c] = (bf16_t)v;
}

// ---------------------------------------------------------------------------
// Edge relu-sum: R[n] = sum_{i<eTot} relu(Hs[src] + Hd[n] + b_m1)
// X1 = [Hs | Hd] (N x 384).  Edges are dst-sorted, eTot (=20) per node.
// ---------------------------------------------------------------------------
__global__ __launch_bounds__(192) void edge_relu_sum(
    const float* __restrict__ X1, const int* __restrict__ srcList,
    const float* __restrict__ b_m1, bf16_t* __restrict__ R_bf, int eTot) {
  int n = blockIdx.x, c = threadIdx.x;
  const int* src = srcList + (size_t)n * eTot;
  float hd = X1[(size_t)n * 384 + HID + c] + b_m1[c];
  float acc = 0.0f;
  for (int i = 0; i < eTot; ++i) {
    float v = X1[(size_t)src[i] * 384 + c] + hd;
    acc += fmaxf(v, 0.0f);
  }
  R_bf[(size_t)n * HID + c] = (bf16_t)acc;
}

// ---------------------------------------------------------------------------
// Fused GRU cell + LayerNorm + per-step logits (h_new @ W_out + b_out).
// gi/gh already carry b_ih/b_hh from the GEMM epilogue.
// ---------------------------------------------------------------------------
__global__ __launch_bounds__(192) void gru_ln_logits(
    const float* __restrict__ gi, const float* __restrict__ gh,
    const float* __restrict__ g_n, const float* __restrict__ be_n,
    const float* __restrict__ W_out, const float* __restrict__ b_out,
    float* __restrict__ h, bf16_t* __restrict__ h_bf,
    float* __restrict__ out_all, float* __restrict__ out_last,
    int nNodes, int t) {
  __shared__ float sh[HID];
  __shared__ float hn_sh[HID];
  int n = blockIdx.x, c = threadIdx.x;
  const float* gin = gi + (size_t)n * 576;
  const float* ghn = gh + (size_t)n * 576;
  float r  = 1.0f / (1.0f + expf(-(gin[c]       + ghn[c])));
  float z  = 1.0f / (1.0f + expf(-(gin[192 + c] + ghn[192 + c])));
  float nn = tanhf(gin[384 + c] + r * ghn[384 + c]);
  float hp = h[(size_t)n * HID + c];
  float hc = (1.0f - z) * nn + z * hp;
  float mean = block_sum192(sh, hc, c) * (1.0f / HID);
  float dv = hc - mean;
  float var = block_sum192(sh, dv * dv, c) * (1.0f / HID);
  float hn2 = dv * rsqrtf(var + 1e-5f) * g_n[c] + be_n[c];
  h[(size_t)n * HID + c] = hn2;
  h_bf[(size_t)n * HID + c] = (bf16_t)hn2;
  hn_sh[c] = hn2;
  __syncthreads();
  if (c < 9) {
    float acc = b_out[c];
    for (int k = 0; k < HID; ++k) acc += hn_sh[k] * W_out[k * 9 + c];
    out_all[((size_t)t * nNodes + n) * 9 + c] = acc;
    if (t == NSTEPS - 1) out_last[(size_t)n * 9 + c] = acc;
  }
}

// ---------------------------------------------------------------------------
extern "C" void kernel_launch(void* const* d_in, const int* in_sizes, int n_in,
                              void* d_out, int out_size, void* d_ws, size_t ws_size,
                              hipStream_t stream) {
  const float* x     = (const float*)d_in[0];
  const int*   edge  = (const int*)d_in[1];
  const float* W_in  = (const float*)d_in[2];
  const float* b_in  = (const float*)d_in[3];
  const float* g_in  = (const float*)d_in[4];
  const float* be_in = (const float*)d_in[5];
  const float* pos   = (const float*)d_in[6];
  const float* W_m1  = (const float*)d_in[7];
  const float* b_m1  = (const float*)d_in[8];
  const float* W_m2  = (const float*)d_in[9];
  const float* b_m2  = (const float*)d_in[10];
  const float* W_ih  = (const float*)d_in[11];
  const float* W_hh  = (const float*)d_in[12];
  const float* b_ih  = (const float*)d_in[13];
  const float* b_hh  = (const float*)d_in[14];
  const float* g_n   = (const float*)d_in[15];
  const float* be_n  = (const float*)d_in[16];
  const float* W_out = (const float*)d_in[17];
  const float* b_out = (const float*)d_in[18];

  const int N = in_sizes[0] / 10;          // 5184 nodes
  const int E = in_sizes[1] / 2;           // 103680 edges
  const int ePerNode = E / N;              // 20 (dst-sorted edge list)
  const int* srcList = edge + E;           // edge_index[1]

  // ---- workspace carve (256B aligned) ----
  char* ws = (char*)d_ws;
  auto carve = [&](size_t bytes) -> void* {
    void* p = (void*)ws;
    ws += (bytes + 255) & ~(size_t)255;
    return p;
  };
  float*  h      = (float*)carve((size_t)N * HID * 4);
  bf16_t* h_bf   = (bf16_t*)carve((size_t)N * HID * 2);
  float*  X1     = (float*)carve((size_t)N * 384 * 4);   // [Hs | Hd]
  bf16_t* R_bf   = (bf16_t*)carve((size_t)N * HID * 2);
  bf16_t* agg_bf = (bf16_t*)carve((size_t)N * HID * 2);
  float*  gi     = (float*)carve((size_t)N * 576 * 4);
  float*  gh     = (float*)carve((size_t)N * 576 * 4);
  bf16_t* WT_m1  = (bf16_t*)carve((size_t)384 * HID * 2);
  bf16_t* WT_m2  = (bf16_t*)carve((size_t)HID * HID * 2);
  bf16_t* Wih_bf = (bf16_t*)carve((size_t)576 * HID * 2);
  bf16_t* Whh_bf = (bf16_t*)carve((size_t)576 * HID * 2);
  float*  b_m2s  = (float*)carve((size_t)HID * 4);

  float* out_last = (float*)d_out;                 // (N, 9)
  float* out_all  = (float*)d_out + (size_t)N * 9; // (16, N, 9)

  // ---- weight prep (bf16 + transpose) ----
  prep_weights<<<(576 * HID + 255) / 256, 256, 0, stream>>>(
      W_m1, W_m2, W_ih, W_hh, b_m2, WT_m1, WT_m2, Wih_bf, Whh_bf, b_m2s);

  // ---- h0 ----
  init_h<<<N, HID, 0, stream>>>(x, W_in, b_in, g_in, be_in, pos, h, h_bf);

  const int waveStrips = N / (16 * RS);              // 162 strips of 32 rows
  const int rb = (waveStrips + 7) / 8;               // 21 row blocks

  for (int t = 0; t < NSTEPS; ++t) {
    // X1 = h @ [W_m1_top | W_m1_bot]  -> (N, 384) = [Hs | Hd]
    gemm_k192_wmma<false, true, false><<<dim3(rb, 384 / GCOLS), 256, 0, stream>>>(
        h_bf, WT_m1, nullptr, X1, nullptr, N, 384);
    // R[n] = sum_edges relu(Hs[src] + Hd[n] + b_m1)
    edge_relu_sum<<<N, HID, 0, stream>>>(X1, srcList, b_m1, R_bf, ePerNode);
    // agg = R @ W_m2 + 20*b_m2   (sum commutes with the linear layer)
    gemm_k192_wmma<true, false, true><<<dim3(rb, HID / GCOLS), 256, 0, stream>>>(
        R_bf, WT_m2, b_m2s, nullptr, agg_bf, N, HID);
    // gi = agg @ W_ih^T + b_ih ; gh = h @ W_hh^T + b_hh
    gemm_k192_wmma<true, true, false><<<dim3(rb, 576 / GCOLS), 256, 0, stream>>>(
        agg_bf, Wih_bf, b_ih, gi, nullptr, N, 576);
    gemm_k192_wmma<true, true, false><<<dim3(rb, 576 / GCOLS), 256, 0, stream>>>(
        h_bf, Whh_bf, b_hh, gh, nullptr, N, 576);
    // GRU + LN + logits
    gru_ln_logits<<<N, HID, 0, stream>>>(
        gi, gh, g_n, be_n, W_out, b_out, h, h_bf, out_all, out_last, N, t);
  }
}
